// TurboQuantAttention_88364657148620
// MI455X (gfx1250) — matrix operand order repaired and checked by reading, not compiled, not run on hardware
//
#include <hip/hip_runtime.h>
#include <math.h>

typedef __attribute__((ext_vector_type(16))) _Float16 v16h;
typedef __attribute__((ext_vector_type(8)))  _Float16 v8h;
typedef __attribute__((ext_vector_type(8)))  float    v8f;

#define B_ 2
#define H_ 16
#define S_ 2048
#define D_ 128
#define NROW (B_*H_*S_)           // 65536 rows per tensor

#define HALF_RANGE_ 3.5f
#define INCREMENT_  0.25f

#define KSTRIDE 136               // halves; 272B = 68 dwords -> conflict-free
#define VSTRIDE 40                // halves;  80B = 20 dwords -> conflict-free

// ---------------------------------------------------------------------------
// Kernel 0: zero the two abs-max accumulators (ws is poisoned, must self-init)
// ---------------------------------------------------------------------------
__global__ void tqa_zero_max(unsigned int* maxes) {
  if (threadIdx.x < 2) maxes[threadIdx.x] = 0u;
}

// ---------------------------------------------------------------------------
// Kernel 1: f16 copies of R (row-major) and R^T; 32KB each, L2-resident.
// ---------------------------------------------------------------------------
__global__ void tqa_prep_R(const float* __restrict__ R,
                           _Float16* __restrict__ r16,
                           _Float16* __restrict__ r16t) {
  int i = blockIdx.x * blockDim.x + threadIdx.x;
  if (i < D_ * D_) {
    float val = R[i];
    r16[i] = (_Float16)val;
    int d = i >> 7, e = i & (D_ - 1);
    r16t[e * D_ + d] = (_Float16)val;       // R^T[e][d] = R[d][e]
  }
}

// ---------------------------------------------------------------------------
// Kernel 2: xr = x @ R on the matrix pipe (f16 in, f32 accumulate),
// plus global abs-max: per-lane -> wave shfl reduce -> one atomic per block.
//   grid = (NROW/64, 2); 128 threads = 4 waves; each wave = 16 rows x 128 cols.
// WMMA layouts (CDNA5 ISA):
//   A 16x32 f16 : lane L -> row L%16; half h -> K = 32c + (L/16)*16 + h (contiguous)
//   B 32x16 f16 : lane L -> col L%16; half h -> K = (L/16)*16 + h (contiguous)
//   C   16x16   : lane L -> col L%16; VGPR r -> row r + 8*(L/16)
// ---------------------------------------------------------------------------
__global__ __launch_bounds__(128)
void tqa_rotate_fwd_wmma(const float* __restrict__ k, const float* __restrict__ v,
                         const _Float16* __restrict__ R16T,
                         float* __restrict__ kr, float* __restrict__ vr,
                         unsigned int* __restrict__ maxes) {
  __shared__ float wmax[4];
  const int tid  = threadIdx.x;
  const int wave = tid >> 5, lane = tid & 31, hi = lane >> 4, ln = lane & 15;
  const int which = blockIdx.y;
  const float* src = which ? v  : k;
  float*       dst = which ? vr : kr;
  const size_t row0 = (size_t)blockIdx.x * 64 + wave * 16;

  // A-fragments: x rows, f32 -> f16 on the fly (loads land at A-layout offsets)
  v16h a[4];
  {
    const float* xrow = src + (row0 + ln) * D_;
#pragma unroll
    for (int c = 0; c < 4; ++c) {
      v8f f0 = *(const v8f*)(xrow + 32 * c + hi * 16);
      v8f f1 = *(const v8f*)(xrow + 32 * c + hi * 16 + 8);
      union { v16h v; v8h h[2]; } u;
#pragma unroll
      for (int j = 0; j < 8; ++j) {
        u.h[0][j] = (_Float16)f0[j];
        u.h[1][j] = (_Float16)f1[j];
      }
      a[c] = u.v;
    }
  }

  v8f acc[8];
  const v8f z = {};
#pragma unroll
  for (int i = 0; i < 8; ++i) acc[i] = z;

  // xr(16x128) = A(16x128) x R(128x128): B-frags are contiguous rows of R^T
#pragma unroll
  for (int c = 0; c < 4; ++c) {
#pragma unroll
    for (int ch = 0; ch < 8; ++ch) {
      v16h b = *(const v16h*)(R16T + (size_t)(ch * 16 + ln) * D_ + 32 * c + hi * 16);
      acc[ch] = __builtin_amdgcn_wmma_f32_16x16x32_f16(false, a[c], false, b,
                                                       (short)0, acc[ch], false, false);
    }
  }

  float lm = 0.f;
#pragma unroll
  for (int ch = 0; ch < 8; ++ch) {
#pragma unroll
    for (int r = 0; r < 8; ++r) {
      float val = acc[ch][r];
      dst[(row0 + r + 8 * hi) * D_ + ch * 16 + ln] = val;
      lm = fmaxf(lm, fabsf(val));
    }
  }
#pragma unroll
  for (int m = 16; m >= 1; m >>= 1) lm = fmaxf(lm, __shfl_xor(lm, m, 32));
  if (lane == 0) wmax[wave] = lm;
  __syncthreads();
  if (tid == 0) {
    float bm = fmaxf(fmaxf(wmax[0], wmax[1]), fmaxf(wmax[2], wmax[3]));
    atomicMax(&maxes[which], __float_as_uint(bm));
  }
}

// ---------------------------------------------------------------------------
// Kernel 3: 3-bit quant + QJL sign correction + dequant applied while building
// the A-fragment, then inverse rotation xd @ R^T on the matrix pipe -> f16.
// B-frags for R^T columns are contiguous rows of R (row-major f16).
// ---------------------------------------------------------------------------
__device__ __forceinline__ float tqa_qd(float x, float toS, float fromS, float corr) {
  float xs = x * toS;                               // scale to [-3.5, 3.5]
  float xq = rintf(xs * 4.f) * INCREMENT_;          // round-half-even buckets
  xq = fminf(fmaxf(xq, -HALF_RANGE_), HALF_RANGE_);
  float resid = xs - xq;
  float sgn = (resid > 0.f) ? 1.f : ((resid < 0.f) ? -1.f : 0.f);
  return (xq + sgn * corr) * fromS;                 // dequantize
}

__global__ __launch_bounds__(128)
void tqa_quant_invrot_wmma(const float* __restrict__ kr, const float* __restrict__ vr,
                           const _Float16* __restrict__ R16,
                           const float* __restrict__ qjl,
                           const unsigned int* __restrict__ maxes,
                           _Float16* __restrict__ k16, _Float16* __restrict__ v16) {
  const int tid  = threadIdx.x;
  const int wave = tid >> 5, lane = tid & 31, hi = lane >> 4, ln = lane & 15;
  const int which = blockIdx.y;
  const float* src = which ? vr  : kr;
  _Float16*    dst = which ? v16 : k16;
  const size_t row0 = (size_t)blockIdx.x * 64 + wave * 16;

  const float xmax  = __uint_as_float(maxes[which]) + 1e-8f;
  const float toS   = HALF_RANGE_ / xmax;
  const float fromS = xmax / HALF_RANGE_;
  const float corr  = qjl[0] * xmax;

  // A-fragments: quant/dequant element-wise, cvt to f16
  v16h a[4];
  {
    const float* xrow = src + (row0 + ln) * D_;
#pragma unroll
    for (int c = 0; c < 4; ++c) {
      v8f f0 = *(const v8f*)(xrow + 32 * c + hi * 16);
      v8f f1 = *(const v8f*)(xrow + 32 * c + hi * 16 + 8);
      union { v16h v; v8h h[2]; } u;
#pragma unroll
      for (int j = 0; j < 8; ++j) {
        u.h[0][j] = (_Float16)tqa_qd(f0[j], toS, fromS, corr);
        u.h[1][j] = (_Float16)tqa_qd(f1[j], toS, fromS, corr);
      }
      a[c] = u.v;
    }
  }

  v8f acc[8];
  const v8f z = {};
#pragma unroll
  for (int i = 0; i < 8; ++i) acc[i] = z;

  // xo(16x128) = xd(16x128) x R^T(128x128): B(e, d) = R[d][e] -> rows of R16
#pragma unroll
  for (int c = 0; c < 4; ++c) {
#pragma unroll
    for (int ch = 0; ch < 8; ++ch) {
      v16h b = *(const v16h*)(R16 + (size_t)(ch * 16 + ln) * D_ + 32 * c + hi * 16);
      acc[ch] = __builtin_amdgcn_wmma_f32_16x16x32_f16(false, a[c], false, b,
                                                       (short)0, acc[ch], false, false);
    }
  }

#pragma unroll
  for (int ch = 0; ch < 8; ++ch) {
#pragma unroll
    for (int r = 0; r < 8; ++r)
      dst[(row0 + r + 8 * hi) * D_ + ch * 16 + ln] = (_Float16)acc[ch][r];
  }
}

// ---------------------------------------------------------------------------
// Kernel 4: Q f32 -> f16
// ---------------------------------------------------------------------------
__global__ void tqa_cvt_q(const float* __restrict__ q, _Float16* __restrict__ q16, int n) {
  int i = blockIdx.x * blockDim.x + threadIdx.x;
  if (i < n) q16[i] = (_Float16)q[i];
}

// ---------------------------------------------------------------------------
// Kernel 5: WMMA flash attention, double-buffered LDS staging of K and V^T.
// ---------------------------------------------------------------------------
struct TileRegs { v8h k[4]; v8h v[4]; };

__device__ __forceinline__
void tqa_load_tile(const _Float16* __restrict__ K, const _Float16* __restrict__ V,
                   size_t base, int kv, int tid, TileRegs& t) {
  const int r  = tid & 31;
  const int g4 = tid >> 5;
  const _Float16* krow = K + base + (size_t)(kv + r) * D_ + g4 * 32;
  const _Float16* vrow = V + base + (size_t)(kv + r) * D_ + g4 * 32;
#pragma unroll
  for (int j = 0; j < 4; ++j) {
    t.k[j] = *(const v8h*)(krow + 8 * j);
    t.v[j] = *(const v8h*)(vrow + 8 * j);
  }
}

__device__ __forceinline__
void tqa_store_tile(_Float16* __restrict__ ks, _Float16* __restrict__ vsT,
                    int tid, const TileRegs& t) {
  const int r  = tid & 31;
  const int g4 = tid >> 5;
#pragma unroll
  for (int j = 0; j < 4; ++j)
    *(v8h*)(ks + r * KSTRIDE + g4 * 32 + 8 * j) = t.k[j];
  union { v8h v[4]; _Float16 h[32]; } u;
#pragma unroll
  for (int j = 0; j < 4; ++j) u.v[j] = t.v[j];
#pragma unroll
  for (int e = 0; e < 32; ++e)
    vsT[(g4 * 32 + e) * VSTRIDE + r] = u.h[e];
}

__global__ __launch_bounds__(128)
void tqa_flash_attn(const _Float16* __restrict__ Q,
                    const _Float16* __restrict__ K,
                    const _Float16* __restrict__ V,
                    float* __restrict__ O) {
  __shared__ _Float16 KS[2][32 * KSTRIDE];
  __shared__ _Float16 VS[2][D_ * VSTRIDE];
  __shared__ _Float16 PB[4][16 * 32];

  const int tid  = threadIdx.x;
  const int wave = tid >> 5;
  const int lane = tid & 31;
  const int hi   = lane >> 4;
  const int ln   = lane & 15;

  const int    qtile = blockIdx.x;
  const int    bh    = blockIdx.y;
  const size_t base  = (size_t)bh * S_ * D_;
  const int    q0    = qtile * 64 + wave * 16;

  v16h aq[4];
  {
    const _Float16* qrow = Q + base + (size_t)(q0 + ln) * D_;
#pragma unroll
    for (int c = 0; c < 4; ++c) {
      union { v16h v; v8h h[2]; } u;
      u.h[0] = *(const v8h*)(qrow + 32 * c + hi * 16);
      u.h[1] = *(const v8h*)(qrow + 32 * c + hi * 16 + 8);
      aq[c] = u.v;
    }
  }

  v8f o[8];
  const v8f vzero = {};
#pragma unroll
  for (int i = 0; i < 8; ++i) o[i] = vzero;

  float mrow[8], lrow[8];
#pragma unroll
  for (int r = 0; r < 8; ++r) { mrow[r] = -INFINITY; lrow[r] = 0.f; }

  const float scale = 0.08838834764831844f;   // 1/sqrt(128)
  _Float16* pb = PB[wave];

  const int NT = S_ / 32;
  TileRegs tr;
  tqa_load_tile(K, V, base, 0, tid, tr);
  tqa_store_tile(KS[0], VS[0], tid, tr);
  __syncthreads();

  for (int it = 0; it < NT; ++it) {
    const int  cur  = it & 1;
    const bool more = (it + 1) < NT;

    if (more) tqa_load_tile(K, V, base, (it + 1) * 32, tid, tr);
    if (it + 2 < NT)
      __builtin_prefetch(K + base + (size_t)((it + 2) * 32 + (tid & 31)) * D_, 0, 3);

    const _Float16* ks  = KS[cur];
    const _Float16* vsT = VS[cur];

    v8f s0 = vzero, s1 = vzero;
#pragma unroll
    for (int c = 0; c < 4; ++c) {
      union { v16h v; v8h h[2]; } b0, b1;
      const _Float16* k0 = ks + (ln)      * KSTRIDE + 32 * c + hi * 16;
      const _Float16* k1 = ks + (16 + ln) * KSTRIDE + 32 * c + hi * 16;
      b0.h[0] = *(const v8h*)(k0);  b0.h[1] = *(const v8h*)(k0 + 8);
      b1.h[0] = *(const v8h*)(k1);  b1.h[1] = *(const v8h*)(k1 + 8);
      s0 = __builtin_amdgcn_wmma_f32_16x16x32_f16(false, aq[c], false, b0.v,
                                                  (short)0, s0, false, false);
      s1 = __builtin_amdgcn_wmma_f32_16x16x32_f16(false, aq[c], false, b1.v,
                                                  (short)0, s1, false, false);
    }

    float fac[8];
#pragma unroll
    for (int r = 0; r < 8; ++r) {
      float a0 = s0[r] * scale, a1 = s1[r] * scale;
      float t = fmaxf(a0, a1);
#pragma unroll
      for (int m = 8; m >= 1; m >>= 1) t = fmaxf(t, __shfl_xor(t, m, 32));
      float mnew = fmaxf(mrow[r], t);
      fac[r] = __expf(mrow[r] - mnew);
      float p0 = __expf(a0 - mnew);
      float p1 = __expf(a1 - mnew);
      s0[r] = p0; s1[r] = p1;
      float ps = p0 + p1;
#pragma unroll
      for (int m = 8; m >= 1; m >>= 1) ps += __shfl_xor(ps, m, 32);
      lrow[r] = lrow[r] * fac[r] + ps;
      mrow[r] = mnew;
    }

#pragma unroll
    for (int ch = 0; ch < 8; ++ch) {
#pragma unroll
      for (int r = 0; r < 8; ++r) o[ch][r] *= fac[r];
    }

#pragma unroll
    for (int r = 0; r < 8; ++r) {
      pb[(r + 8 * hi) * 32 +      ln] = (_Float16)s0[r];
      pb[(r + 8 * hi) * 32 + 16 + ln] = (_Float16)s1[r];
    }
    asm volatile("s_wait_dscnt 0" ::: "memory");   // wave-local LDS RAW

    union { v16h v; v8h h[2]; } up;
    up.h[0] = *(const v8h*)(pb + ln * 32 + hi * 16);
    up.h[1] = *(const v8h*)(pb + ln * 32 + hi * 16 + 8);

#pragma unroll
    for (int ch = 0; ch < 8; ++ch) {
      union { v16h v; v8h h[2]; } bv;
      const _Float16* vp = vsT + (ch * 16 + ln) * VSTRIDE + hi * 16;
      bv.h[0] = *(const v8h*)(vp);
      bv.h[1] = *(const v8h*)(vp + 8);
      o[ch] = __builtin_amdgcn_wmma_f32_16x16x32_f16(false, up.v, false, bv.v,
                                                     (short)0, o[ch], false, false);
    }

    if (more) tqa_store_tile(KS[cur ^ 1], VS[cur ^ 1], tid, tr);
    __syncthreads();
  }

#pragma unroll
  for (int r = 0; r < 8; ++r) {
    float inv = 1.0f / lrow[r];
    float* orow = O + base + (size_t)(q0 + r + 8 * hi) * D_;
#pragma unroll
    for (int ch = 0; ch < 8; ++ch)
      orow[ch * 16 + ln] = o[ch][r] * inv;
  }
}

// ---------------------------------------------------------------------------
extern "C" void kernel_launch(void* const* d_in, const int* in_sizes, int n_in,
                              void* d_out, int out_size, void* d_ws, size_t ws_size,
                              hipStream_t stream) {
  const float* q   = (const float*)d_in[0];
  const float* k   = (const float*)d_in[1];
  const float* v   = (const float*)d_in[2];
  const float* R   = (const float*)d_in[3];
  const float* qjl = (const float*)d_in[4];

  const size_t nElem = (size_t)NROW * D_;

  char* ws = (char*)d_ws;
  float*    kr   = (float*)ws;        ws += nElem * sizeof(float);
  float*    vr   = (float*)ws;        ws += nElem * sizeof(float);
  _Float16* k16  = (_Float16*)ws;     ws += nElem * sizeof(_Float16);
  _Float16* v16  = (_Float16*)ws;     ws += nElem * sizeof(_Float16);
  _Float16* q16  = (_Float16*)ws;     ws += nElem * sizeof(_Float16);
  _Float16* r16  = (_Float16*)ws;     ws += (size_t)D_ * D_ * sizeof(_Float16);
  _Float16* r16t = (_Float16*)ws;     ws += (size_t)D_ * D_ * sizeof(_Float16);
  unsigned int* maxes = (unsigned int*)ws;

  tqa_zero_max<<<1, 32, 0, stream>>>(maxes);
  tqa_prep_R<<<(D_ * D_ + 255) / 256, 256, 0, stream>>>(R, r16, r16t);
  tqa_rotate_fwd_wmma<<<dim3(NROW / 64, 2), 128, 0, stream>>>(k, v, r16t, kr, vr, maxes);
  tqa_quant_invrot_wmma<<<dim3(NROW / 64, 2), 128, 0, stream>>>(kr, vr, r16, qjl, maxes,
                                                                k16, v16);
  tqa_cvt_q<<<(unsigned)((nElem + 255) / 256), 256, 0, stream>>>(q, q16, (int)nElem);
  tqa_flash_attn<<<dim3(S_ / 64, B_ * H_), 128, 0, stream>>>(q16, k16, v16, (float*)d_out);
}
